// FraudGNN_77687368450593
// MI455X (gfx1250) — compile-verified, matrix operand order checked by reading, not run on hardware
//
#include <hip/hip_runtime.h>
#include <hip/hip_bf16.h>
#include <stdint.h>

#define NEG_SLOPE 0.2f
#define GAT_EPS 1e-16f

typedef __attribute__((ext_vector_type(16))) __bf16 v16bf_t;
typedef __attribute__((ext_vector_type(8)))  float  v8f_t;

union FragAB { uint4 q[2]; v16bf_t v; };

static __device__ __forceinline__ unsigned short f32_to_bf16(float f) {
  unsigned int u = __float_as_uint(f);
  u += 0x7FFFu + ((u >> 16) & 1u);          // round to nearest even
  return (unsigned short)(u >> 16);
}

static __device__ __forceinline__ void atomicMaxF(float* addr, float val) {
  if (val >= 0.f) atomicMax((int*)addr, __float_as_int(val));
  else            atomicMin((unsigned int*)addr, __float_as_uint(val));
}

static __device__ __forceinline__ void edge_sd(const long long* __restrict__ ei,
                                               long long E, long long e,
                                               int& s, int& d) {
  if (e < E) { s = (int)ei[e]; d = (int)ei[E + e]; }
  else       { int v = (int)(e - E); s = v; d = v; }   // self-loop
}

// ---------------------------------------------------------------- conversions
__global__ void k_f32_to_bf16(const float* __restrict__ in,
                              unsigned short* __restrict__ out, long long n) {
  long long i = (long long)blockIdx.x * blockDim.x + threadIdx.x;
  if (i < n) out[i] = f32_to_bf16(in[i]);
}

__global__ void k_fill_f32(float* __restrict__ p, float v, long long n) {
  long long i = (long long)blockIdx.x * blockDim.x + threadIdx.x;
  if (i < n) p[i] = v;
}

// ---------------------------------------------------------------- WMMA GEMM
// C[M x Nc] = A[M x K] (bf16, row-major) * B[Nc x K] (bf16, row-major)^T
// One wave computes one 16x16 C tile via v_wmma_f32_16x16x32_bf16.
// Per-lane fragment K pattern (16-bit A/B layout, ISA 7.12.2):
//   lanes 0..15 : K = kb+{0..7, 16..23},  row/col = lane
//   lanes 16..31: K = kb+{8..15,24..31},  row/col = lane-16
__global__ void k_wmma_gemm_bf16(const unsigned short* __restrict__ A,
                                 const unsigned short* __restrict__ B,
                                 float* __restrict__ C,
                                 int M, int Nc, int K) {
  const int lane  = threadIdx.x;                       // 0..31 (wave32)
  const int tileN = blockIdx.y * blockDim.y + threadIdx.y;
  const int m0 = blockIdx.x * 16;
  const int n0 = tileN * 16;
  if (n0 >= Nc) return;                                // wave-uniform

  const int lm = lane & 15;
  const int hi = lane >> 4;
  const int kf = hi * 8;

  int ar = m0 + lm; if (ar >= M)  ar = M - 1;
  int br = n0 + lm; if (br >= Nc) br = Nc - 1;
  const unsigned short* ap = A + (size_t)ar * K;
  const unsigned short* bp = B + (size_t)br * K;       // B[k][n] = Bmat[n][k]

  v8f_t acc = {};
  for (int kb = 0; kb < K; kb += 32) {
    FragAB a, b;
    a.q[0] = *(const uint4*)(ap + kb + kf);
    a.q[1] = *(const uint4*)(ap + kb + kf + 16);
    b.q[0] = *(const uint4*)(bp + kb + kf);
    b.q[1] = *(const uint4*)(bp + kb + kf + 16);
    if (kb + 32 < K) {
      __builtin_prefetch(ap + kb + 32 + kf, 0, 1);     // global_prefetch_b8
      __builtin_prefetch(bp + kb + 32 + kf, 0, 1);
    }
    acc = __builtin_amdgcn_wmma_f32_16x16x32_bf16(
        false, a.v, false, b.v, (short)0, acc, false, false);
  }

  // C/D layout: VGPR r -> M = hi*8 + r, N = lane%16
  const int n  = n0 + lm;
  const int mb = m0 + hi * 8;
#pragma unroll
  for (int r = 0; r < 8; ++r) {
    int m = mb + r;
    if (m < M && n < Nc) C[(size_t)m * Nc + n] = acc[r];
  }
}

// ------------------------------------------------------- attention logits
// out[n*H+h] = dot(xt[n,h,:], att[h,:])   (xt row (n*H+h) is contiguous len C)
__global__ void k_att_dot(const float* __restrict__ xt, const float* __restrict__ att,
                          float* __restrict__ outv, int Nn, int H, int C) {
  long long i = (long long)blockIdx.x * blockDim.x + threadIdx.x;
  if (i >= (long long)Nn * H) return;
  int h = (int)(i % H);
  const float* xp = xt + (size_t)i * C;
  const float* ap = att + (size_t)h * C;
  float s = 0.f;
  for (int c = 0; c < C; ++c) s += xp[c] * ap[c];
  outv[i] = s;
}

// ------------------------------------------------------- segment softmax
__global__ void k_edge_logit_max(const long long* __restrict__ ei, long long E,
                                 const float* __restrict__ asrc,
                                 const float* __restrict__ adst,
                                 float* __restrict__ ew, float* __restrict__ mmax,
                                 int H, long long Ep) {
  long long i = (long long)blockIdx.x * blockDim.x + threadIdx.x;
  if (i >= Ep * H) return;
  long long e = i / H; int h = (int)(i - e * H);
  int s, d; edge_sd(ei, E, e, s, d);
  float l = asrc[(size_t)s * H + h] + adst[(size_t)d * H + h];
  l = l > 0.f ? l : NEG_SLOPE * l;                     // LeakyReLU(0.2)
  ew[i] = l;
  atomicMaxF(&mmax[(size_t)d * H + h], l);
}

__global__ void k_edge_exp_sum(const long long* __restrict__ ei, long long E,
                               const float* __restrict__ mmax,
                               float* __restrict__ ew, float* __restrict__ ssum,
                               int H, long long Ep) {
  long long i = (long long)blockIdx.x * blockDim.x + threadIdx.x;
  if (i >= Ep * H) return;
  long long e = i / H; int h = (int)(i - e * H);
  int s, d; edge_sd(ei, E, e, s, d); (void)s;
  float v = __expf(ew[i] - mmax[(size_t)d * H + h]);
  ew[i] = v;
  atomicAdd(&ssum[(size_t)d * H + h], v);
}

__global__ void k_edge_message(const long long* __restrict__ ei, long long E,
                               const float* __restrict__ xt,
                               const float* __restrict__ ew,
                               const float* __restrict__ ssum,
                               float* __restrict__ acc,
                               int H, int C, long long Ep) {
  long long i = (long long)blockIdx.x * blockDim.x + threadIdx.x;
  if (i >= Ep * (long long)(H * C)) return;
  long long e = i / (H * C);
  int rem = (int)(i - e * (H * C));
  int h = rem / C, c = rem - h * C;
  int s, d; edge_sd(ei, E, e, s, d);
  float alpha = ew[e * H + h] / (ssum[(size_t)d * H + h] + GAT_EPS);
  float val = xt[((size_t)s * H + h) * C + c] * alpha;
  atomicAdd(&acc[((size_t)d * H + h) * C + c], val);
}

// ------------------------------------------------------- epilogues
__global__ void k_bias_relu_bf16(const float* __restrict__ acc,
                                 const float* __restrict__ bias,
                                 unsigned short* __restrict__ outbf,
                                 long long total, int F) {
  long long i = (long long)blockIdx.x * blockDim.x + threadIdx.x;
  if (i >= total) return;
  int c = (int)(i % F);
  float v = acc[i] + bias[c];
  outbf[i] = f32_to_bf16(v > 0.f ? v : 0.f);
}

__global__ void k_final(const float* __restrict__ acc2, const float* __restrict__ b2,
                        const float* __restrict__ fcw, const float* __restrict__ fcb,
                        float* __restrict__ out, int Nn, int C) {
  int n = blockIdx.x * blockDim.x + threadIdx.x;
  if (n >= Nn) return;
  float s = 0.f;
  for (int c = 0; c < C; ++c) {
    float v = acc2[(size_t)n * C + c] + b2[c];
    v = v > 0.f ? v : 0.f;                             // ReLU of layer 2
    s += v * fcw[c];
  }
  s += fcb[0];
  out[n] = 1.f / (1.f + __expf(-s));                   // sigmoid
}

// ---------------------------------------------------------------- launch
extern "C" void kernel_launch(void* const* d_in, const int* in_sizes, int n_in,
                              void* d_out, int out_size, void* d_ws, size_t ws_size,
                              hipStream_t stream) {
  (void)n_in; (void)out_size; (void)ws_size;
  const float*     x   = (const float*)d_in[0];
  const long long* ei  = (const long long*)d_in[1];     // int64 [2,E]
  const float*     W1  = (const float*)d_in[2];
  const float*     as1 = (const float*)d_in[3];
  const float*     ad1 = (const float*)d_in[4];
  const float*     b1  = (const float*)d_in[5];
  const float*     W2  = (const float*)d_in[6];
  const float*     as2 = (const float*)d_in[7];
  const float*     ad2 = (const float*)d_in[8];
  const float*     b2  = (const float*)d_in[9];
  const float*     fcw = (const float*)d_in[10];
  const float*     fcb = (const float*)d_in[11];
  float* out = (float*)d_out;

  const int IN = 128, H1 = 4, C1 = 64, F1 = 256, C2 = 64;
  const int Nn = in_sizes[0] / IN;
  const long long E  = (long long)in_sizes[1] / 2;
  const long long Ep = E + Nn;

  // workspace carve-up (256B aligned slabs)
  char* wsb = (char*)d_ws;
  size_t off = 0;
  auto alloc = [&](size_t bytes) -> char* {
    char* p = wsb + off;
    off = (off + bytes + 255) & ~(size_t)255;
    return p;
  };
  unsigned short* xbf   = (unsigned short*)alloc((size_t)Nn * IN * 2);
  unsigned short* w1bf  = (unsigned short*)alloc((size_t)F1 * IN * 2);
  unsigned short* w2bf  = (unsigned short*)alloc((size_t)C2 * F1 * 2);
  float* xt1   = (float*)alloc((size_t)Nn * F1 * 4);
  float* asrc1 = (float*)alloc((size_t)Nn * H1 * 4);
  float* adst1 = (float*)alloc((size_t)Nn * H1 * 4);
  float* m1    = (float*)alloc((size_t)Nn * H1 * 4);
  float* s1    = (float*)alloc((size_t)Nn * H1 * 4);
  float* ew1   = (float*)alloc((size_t)Ep * H1 * 4);
  float* acc1  = (float*)alloc((size_t)Nn * F1 * 4);
  unsigned short* h1bf = (unsigned short*)alloc((size_t)Nn * F1 * 2);
  float* xt2   = (float*)alloc((size_t)Nn * C2 * 4);
  float* asrc2 = (float*)alloc((size_t)Nn * 4);
  float* adst2 = (float*)alloc((size_t)Nn * 4);
  float* m2    = (float*)alloc((size_t)Nn * 4);
  float* s2    = (float*)alloc((size_t)Nn * 4);
  float* ew2   = (float*)alloc((size_t)Ep * 4);
  float* acc2  = (float*)alloc((size_t)Nn * C2 * 4);

  const int TB = 256;
  auto nb = [](long long n, int tb) { return (unsigned)((n + tb - 1) / tb); };
  const float NEG_INF = -3.402823466e38f;

  // bf16 conversions
  k_f32_to_bf16<<<nb((long long)Nn * IN, TB), TB, 0, stream>>>(x, xbf, (long long)Nn * IN);
  k_f32_to_bf16<<<nb((long long)F1 * IN, TB), TB, 0, stream>>>(W1, w1bf, (long long)F1 * IN);
  k_f32_to_bf16<<<nb((long long)C2 * F1, TB), TB, 0, stream>>>(W2, w2bf, (long long)C2 * F1);

  // ---- layer 1: xt1 = x @ W1^T  (WMMA bf16)
  {
    dim3 blk(32, 4), grd((Nn + 15) / 16, (F1 / 16 + 3) / 4);
    k_wmma_gemm_bf16<<<grd, blk, 0, stream>>>(xbf, w1bf, xt1, Nn, F1, IN);
  }
  k_att_dot<<<nb((long long)Nn * H1, TB), TB, 0, stream>>>(xt1, as1, asrc1, Nn, H1, C1);
  k_att_dot<<<nb((long long)Nn * H1, TB), TB, 0, stream>>>(xt1, ad1, adst1, Nn, H1, C1);
  k_fill_f32<<<nb((long long)Nn * H1, TB), TB, 0, stream>>>(m1, NEG_INF, (long long)Nn * H1);
  k_edge_logit_max<<<nb(Ep * H1, TB), TB, 0, stream>>>(ei, E, asrc1, adst1, ew1, m1, H1, Ep);
  k_fill_f32<<<nb((long long)Nn * H1, TB), TB, 0, stream>>>(s1, 0.f, (long long)Nn * H1);
  k_fill_f32<<<nb((long long)Nn * F1, TB), TB, 0, stream>>>(acc1, 0.f, (long long)Nn * F1);
  k_edge_exp_sum<<<nb(Ep * H1, TB), TB, 0, stream>>>(ei, E, m1, ew1, s1, H1, Ep);
  k_edge_message<<<nb(Ep * H1 * C1, TB), TB, 0, stream>>>(ei, E, xt1, ew1, s1, acc1, H1, C1, Ep);
  k_bias_relu_bf16<<<nb((long long)Nn * F1, TB), TB, 0, stream>>>(acc1, b1, h1bf, (long long)Nn * F1, F1);

  // ---- layer 2: xt2 = h1 @ W2^T  (WMMA bf16)
  {
    dim3 blk(32, 4), grd((Nn + 15) / 16, 1);           // 4 N-tiles via threadIdx.y
    k_wmma_gemm_bf16<<<grd, blk, 0, stream>>>(h1bf, w2bf, xt2, Nn, C2, F1);
  }
  k_att_dot<<<nb((long long)Nn, TB), TB, 0, stream>>>(xt2, as2, asrc2, Nn, 1, C2);
  k_att_dot<<<nb((long long)Nn, TB), TB, 0, stream>>>(xt2, ad2, adst2, Nn, 1, C2);
  k_fill_f32<<<nb((long long)Nn, TB), TB, 0, stream>>>(m2, NEG_INF, (long long)Nn);
  k_edge_logit_max<<<nb(Ep, TB), TB, 0, stream>>>(ei, E, asrc2, adst2, ew2, m2, 1, Ep);
  k_fill_f32<<<nb((long long)Nn, TB), TB, 0, stream>>>(s2, 0.f, (long long)Nn);
  k_fill_f32<<<nb((long long)Nn * C2, TB), TB, 0, stream>>>(acc2, 0.f, (long long)Nn * C2);
  k_edge_exp_sum<<<nb(Ep, TB), TB, 0, stream>>>(ei, E, m2, ew2, s2, 1, Ep);
  k_edge_message<<<nb(Ep * C2, TB), TB, 0, stream>>>(ei, E, xt2, ew2, s2, acc2, 1, C2, Ep);

  // ---- ReLU + Linear(64->1) + sigmoid
  k_final<<<nb((long long)Nn, TB), TB, 0, stream>>>(acc2, b2, fcw, fcb, out, Nn, C2);
}
